// AreaAttention_77000173683101
// MI455X (gfx1250) — matrix operand order repaired and checked by reading, not compiled, not run on hardware
//
#include <hip/hip_runtime.h>
#include <math.h>

typedef __attribute__((ext_vector_type(16))) _Float16 v16h;
typedef __attribute__((ext_vector_type(8)))  float    v8f;
typedef __attribute__((ext_vector_type(4)))  unsigned int u32x4;
typedef __attribute__((ext_vector_type(8)))  int          i32x8;
typedef __attribute__((ext_vector_type(4)))  int          i32x4;

#define NPIX 4096
#define CCH  128

#if defined(__has_builtin)
#if __has_builtin(__builtin_amdgcn_tensor_load_to_lds) && \
    __has_builtin(__builtin_amdgcn_s_wait_tensorcnt)
#define USE_TDM 1
#endif
#endif
#ifndef USE_TDM
#define USE_TDM 0
#endif

__device__ __forceinline__ v8f wmma_f16(v16h a, v16h b, v8f c) {
    // D = A(16x32 f16) * B(32x16 f16) + C(16x16 f32)
    return __builtin_amdgcn_wmma_f32_16x16x32_f16(false, a, false, b,
                                                  (short)0, c, false, false);
}

// A-fragment K offset for VGPR-pair t (0..7), lane group g (0/1).
__device__ __forceinline__ int afrag_k0(int t, int g) {
    return (t < 4) ? (2 * t + 8 * g) : (16 + 2 * (t - 4) + 8 * g);
}

// Build an A-fragment from a row pointer (fixed M per lane, K varies).
__device__ __forceinline__ v16h load_afrag(const _Float16* row, int g) {
    v16h a;
#pragma unroll
    for (int t = 0; t < 8; ++t) {
        const int k0 = afrag_k0(t, g);
        a[2 * t]     = row[k0];
        a[2 * t + 1] = row[k0 + 1];
    }
    return a;
}

#if USE_TDM
// Issue a TDM 2D tile load (Global -> LDS). Descriptor per ISA 08 §8.3/8.4:
// group0: count=1, lds_addr, global_addr[56:0], type=2 ("image").
// group1: data_size=2B, tensor_dim0/1, tile_dim0/1, tensor_dim0_stride.
// Dims/strides in elements (f16). Fill order: X fastest, then Y -> LDS[y][x].
__device__ __forceinline__ void tdm_load_2d(unsigned lds_off, const void* gaddr,
                                            unsigned dim0, unsigned dim1,
                                            unsigned tile0, unsigned tile1,
                                            unsigned stride0) {
    unsigned long long ga = (unsigned long long)(uintptr_t)gaddr;
    u32x4 g0;
    g0[0] = 1u;                                    // count=1, no gather
    g0[1] = lds_off;                               // LDS byte address
    g0[2] = (unsigned)(ga & 0xFFFFFFFFu);          // global_addr[31:0]
    g0[3] = (unsigned)((ga >> 32) & 0x1FFFFFFu)    // global_addr[56:32]
            | (2u << 30);                          // type=2
    i32x8 g1;
    g1[0] = (int)(1u << 16);                       // data_size=2B (code 1)
    g1[1] = (int)((dim0 & 0xFFFFu) << 16);         // tensor_dim0[15:0]
    g1[2] = (int)((dim0 >> 16) | ((dim1 & 0xFFFFu) << 16));
    g1[3] = (int)((dim1 >> 16) | (tile0 << 16));   // tile_dim0
    g1[4] = (int)(tile1 & 0xFFFFu);                // tile_dim1 (tile_dim2=0)
    g1[5] = (int)stride0;                          // tensor_dim0_stride[31:0]
    g1[6] = 0;                                     // stride0[47:32], stride1 lo
    g1[7] = 0;
    i32x4 gz4 = {0, 0, 0, 0};
    i32x8 gz8 = {0, 0, 0, 0, 0, 0, 0, 0};
    __builtin_amdgcn_tensor_load_to_lds(g0, g1, gz4, gz4, gz8, 0);
}
#endif

// ---------------------------------------------------------------------------
// Kernel 0: transpose Wq/Wk/Wv [O][C] f32 -> fp16 [C][O] for WMMA B-fragments
// ---------------------------------------------------------------------------
__global__ void prep_weights(const float* __restrict__ Wq,
                             const float* __restrict__ Wk,
                             const float* __restrict__ Wv,
                             _Float16* __restrict__ WT) {
    int idx = blockIdx.x * blockDim.x + threadIdx.x;
    if (idx >= 3 * CCH * CCH) return;
    int m = idx >> 14;
    int r = idx & 16383;
    int o = r >> 7, c = r & 127;
    const float* W = (m == 0) ? Wq : (m == 1) ? Wk : Wv;
    WT[m * 16384 + c * CCH + o] = (_Float16)W[o * CCH + c];
}

// ---------------------------------------------------------------------------
// Kernel 1: QKV projection via WMMA. Q pre-scaled by (1/sqrt(C))*log2(e).
// ---------------------------------------------------------------------------
__global__ __launch_bounds__(256) void qkv_kernel(
    const float* __restrict__ x,
    const _Float16* __restrict__ WT,
    const float* __restrict__ bq, const float* __restrict__ bk,
    const float* __restrict__ bvp,
    _Float16* __restrict__ Qh, _Float16* __restrict__ Kh,
    _Float16* __restrict__ Vh) {
    __shared__ __attribute__((aligned(32))) _Float16 xT[16][CCH];

    const int b   = blockIdx.y;
    const int n0  = blockIdx.x * 16;
    const int tid = threadIdx.x;

#pragma unroll
    for (int e = 0; e < 8; ++e) {
        int idx = tid + 256 * e;
        int c = idx >> 4, i = idx & 15;
        xT[i][c] = (_Float16)x[((size_t)b * CCH + c) * NPIX + n0 + i];
    }
    __syncthreads();

    const int lane = tid & 31, w = tid >> 5;
    const int g = lane >> 4, ml = lane & 15;
    const int o0 = w * 16;

    v16h afr[4];
#pragma unroll
    for (int kc = 0; kc < 4; ++kc)
        afr[kc] = load_afrag(&xT[ml][kc * 32], g);

    const float qscale = 0.08838834764831845f * 1.4426950408889634f;

#pragma unroll
    for (int mat = 0; mat < 3; ++mat) {
        const _Float16* Wt = WT + mat * 16384;
        const float* bias = (mat == 0) ? bq : (mat == 1) ? bk : bvp;
        const float bval = bias[o0 + ml];
        v8f acc;
#pragma unroll
        for (int r = 0; r < 8; ++r) acc[r] = bval;
#pragma unroll
        for (int kc = 0; kc < 4; ++kc) {
            v16h bf;
#pragma unroll
            for (int s = 0; s < 16; ++s)
                bf[s] = Wt[(size_t)(kc * 32 + 8 * g + (s & 7) + ((s >> 3) << 4)) * CCH + o0 + ml];
            acc = wmma_f16(afr[kc], bf, acc);
        }
        if (mat == 0) {
#pragma unroll
            for (int r = 0; r < 8; ++r)
                Qh[((size_t)b * NPIX + n0 + r + 8 * g) * CCH + o0 + ml] =
                    (_Float16)(acc[r] * qscale);
        } else if (mat == 1) {
#pragma unroll
            for (int r = 0; r < 8; ++r)
                Kh[((size_t)b * CCH + o0 + ml) * NPIX + n0 + r + 8 * g] =
                    (_Float16)acc[r];
        } else {
#pragma unroll
            for (int r = 0; r < 8; ++r)
                Vh[((size_t)b * NPIX + n0 + r + 8 * g) * CCH + o0 + ml] =
                    (_Float16)acc[r];
        }
    }
}

// ---------------------------------------------------------------------------
// Kernel 2: flash attention + residual, double-buffered K/V chunks staged by
// the Tensor Data Mover (fallback: cooperative vector loads).
// ---------------------------------------------------------------------------
__global__ __launch_bounds__(256) void attn_kernel(
    const float* __restrict__ x,
    const _Float16* __restrict__ Qh, const _Float16* __restrict__ Kh,
    const _Float16* __restrict__ Vh,
    float* __restrict__ out) {
    __shared__ __attribute__((aligned(32))) _Float16 ldsK[2][CCH][32];  // 16 KB
    __shared__ __attribute__((aligned(32))) _Float16 ldsV[2][32][CCH];  // 16 KB
    __shared__ __attribute__((aligned(32))) _Float16 ldsP[8][16][32];   //  8 KB

    const int b = blockIdx.y;
    const int tid = threadIdx.x;
    const int lane = tid & 31, w = tid >> 5;
    const int g = lane >> 4, ml = lane & 15;
    const int n0 = (blockIdx.x * 8 + w) * 16;

    const _Float16* Kbase = Kh + (size_t)b * CCH * NPIX;  // [C][N]
    const _Float16* Vbase = Vh + (size_t)b * NPIX * CCH;  // [N][C]

    // Preload this wave's Q A-fragments (already scaled).
    v16h aq[4];
    {
        const _Float16* qrow = Qh + ((size_t)b * NPIX + n0 + ml) * CCH;
#pragma unroll
        for (int kc = 0; kc < 4; ++kc)
            aq[kc] = load_afrag(qrow + kc * 32, g);
    }

    float m_i[8], l_i[8];
    v8f acc[8];
#pragma unroll
    for (int r = 0; r < 8; ++r) { m_i[r] = -1e30f; l_i[r] = 0.0f; }
#pragma unroll
    for (int ct = 0; ct < 8; ++ct)
#pragma unroll
        for (int r = 0; r < 8; ++r) acc[ct][r] = 0.0f;

    // ---- stage chunk 0 into buffer 0 ----
#if USE_TDM
    if (w == 0) {
        tdm_load_2d((unsigned)(uintptr_t)&ldsK[0][0][0], Kbase,
                    NPIX, CCH, 32, CCH, NPIX);   // K tile: [c][key]
        tdm_load_2d((unsigned)(uintptr_t)&ldsV[0][0][0], Vbase,
                    CCH, NPIX, CCH, 32, CCH);    // V tile: [key][c]
    }
#else
#pragma unroll
    for (int e = 0; e < 16; ++e) {
        int idx = tid + 256 * e;
        { int c = idx >> 5, j = idx & 31;
          ldsK[0][c][j] = Kbase[(size_t)c * NPIX + j]; }
        { int k = idx >> 7, c = idx & 127;
          ldsV[0][k][c] = Vbase[(size_t)k * CCH + c]; }
    }
#endif

    const int NCHUNK = NPIX / 32;
    for (int i = 0; i < NCHUNK; ++i) {
        const int m0 = i * 32;
        const int cur = i & 1;
#if USE_TDM
        if (w == 0) __builtin_amdgcn_s_wait_tensorcnt(0);
#endif
        __syncthreads();   // buffer `cur` is now valid for all waves

        // Prefetch next chunk into the other buffer (overlaps with compute).
        if (i + 1 < NCHUNK) {
            const int nxt = cur ^ 1;
            const int mn = m0 + 32;
#if USE_TDM
            if (w == 0) {
                tdm_load_2d((unsigned)(uintptr_t)&ldsK[nxt][0][0], Kbase + mn,
                            NPIX, CCH, 32, CCH, NPIX);
                tdm_load_2d((unsigned)(uintptr_t)&ldsV[nxt][0][0],
                            Vbase + (size_t)mn * CCH, CCH, NPIX, CCH, 32, CCH);
            }
#else
#pragma unroll
            for (int e = 0; e < 16; ++e) {
                int idx = tid + 256 * e;
                { int c = idx >> 5, j = idx & 31;
                  ldsK[nxt][c][j] = Kbase[(size_t)c * NPIX + mn + j]; }
                { int k = idx >> 7, c = idx & 127;
                  ldsV[nxt][k][c] = Vbase[(size_t)(mn + k) * CCH + c]; }
            }
#endif
        }

        // ---- S = Q K : 16 queries x 32 keys ----
        v8f s0, s1;
#pragma unroll
        for (int r = 0; r < 8; ++r) { s0[r] = 0.0f; s1[r] = 0.0f; }
#pragma unroll
        for (int kc = 0; kc < 4; ++kc) {
            v16h bk0, bk1;
#pragma unroll
            for (int s = 0; s < 16; ++s) {
                bk0[s] = ldsK[cur][kc * 32 + 8 * g + (s & 7) + ((s >> 3) << 4)][ml];
                bk1[s] = ldsK[cur][kc * 32 + 8 * g + (s & 7) + ((s >> 3) << 4)][16 + ml];
            }
            s0 = wmma_f16(aq[kc], bk0, s0);
            s1 = wmma_f16(aq[kc], bk1, s1);
        }

        // ---- online softmax (exp2 domain) ----
        float fscale[8];
#pragma unroll
        for (int r = 0; r < 8; ++r) {
            float mx = fmaxf(s0[r], s1[r]);
#pragma unroll
            for (int msk = 1; msk < 16; msk <<= 1)
                mx = fmaxf(mx, __shfl_xor(mx, msk, 32));
            float mnew = fmaxf(m_i[r], mx);
            float f = exp2f(m_i[r] - mnew);
            float p0 = exp2f(s0[r] - mnew);
            float p1 = exp2f(s1[r] - mnew);
            float rs = p0 + p1;
#pragma unroll
            for (int msk = 1; msk < 16; msk <<= 1)
                rs += __shfl_xor(rs, msk, 32);
            l_i[r] = l_i[r] * f + rs;
            m_i[r] = mnew;
            fscale[r] = f;
            s0[r] = p0;
            s1[r] = p1;
        }
#pragma unroll
        for (int ct = 0; ct < 8; ++ct)
#pragma unroll
            for (int r = 0; r < 8; ++r) acc[ct][r] *= fscale[r];

        // P (C/D layout) -> per-wave LDS -> A-fragment layout.
#pragma unroll
        for (int r = 0; r < 8; ++r) {
            ldsP[w][r + 8 * g][ml]      = (_Float16)s0[r];
            ldsP[w][r + 8 * g][16 + ml] = (_Float16)s1[r];
        }
        v16h ap = load_afrag(&ldsP[w][ml][0], g);

        // ---- O += P V ----
#pragma unroll
        for (int ct = 0; ct < 8; ++ct) {
            v16h vb;
#pragma unroll
            for (int s = 0; s < 16; ++s)
                vb[s] = ldsV[cur][8 * g + (s & 7) + ((s >> 3) << 4)][ct * 16 + ml];
            acc[ct] = wmma_f16(ap, vb, acc[ct]);
        }
        __syncthreads();   // all waves done with `cur` before it is re-filled
    }

    // ---- epilogue: normalize, add residual, write [B][C][N] f32 ----
#pragma unroll
    for (int r = 0; r < 8; ++r) {
        float inv = 1.0f / l_i[r];
        int n = n0 + r + 8 * g;
#pragma unroll
        for (int ct = 0; ct < 8; ++ct) {
            int c = ct * 16 + ml;
            size_t idx = ((size_t)b * CCH + c) * NPIX + n;
            out[idx] = x[idx] + acc[ct][r] * inv;
        }
    }
}

// ---------------------------------------------------------------------------
extern "C" void kernel_launch(void* const* d_in, const int* in_sizes, int n_in,
                              void* d_out, int out_size, void* d_ws,
                              size_t ws_size, hipStream_t stream) {
    (void)in_sizes; (void)n_in; (void)out_size; (void)ws_size;
    const float* x   = (const float*)d_in[0];
    const float* Wq  = (const float*)d_in[1];
    const float* bq  = (const float*)d_in[2];
    const float* Wk  = (const float*)d_in[3];
    const float* bk  = (const float*)d_in[4];
    const float* Wv  = (const float*)d_in[5];
    const float* bvp = (const float*)d_in[6];
    float* out = (float*)d_out;

    _Float16* base = (_Float16*)d_ws;
    _Float16* WT = base;
    _Float16* Qh = base + (size_t)3 * CCH * CCH;
    _Float16* Kh = Qh + (size_t)8 * NPIX * CCH;
    _Float16* Vh = Kh + (size_t)8 * NPIX * CCH;

    prep_weights<<<dim3(192), dim3(256), 0, stream>>>(Wq, Wk, Wv, WT);
    qkv_kernel<<<dim3(NPIX / 16, 8), dim3(256), 0, stream>>>(
        x, WT, bq, bk, bvp, Qh, Kh, Vh);
    attn_kernel<<<dim3(NPIX / 16 / 8, 8), dim3(256), 0, stream>>>(
        x, Qh, Kh, Vh, out);
}
